// Model_37580963840523
// MI455X (gfx1250) — matrix operand order repaired and checked by reading, not compiled, not run on hardware
//
#include <hip/hip_runtime.h>
#include <math.h>

// ---------------------------------------------------------------------------
// CDNA5 (gfx1250) wave32 WMMA types
// ---------------------------------------------------------------------------
typedef __attribute__((ext_vector_type(16))) __bf16          v16bf;
typedef __attribute__((ext_vector_type(8)))  float           v8f;
typedef __attribute__((ext_vector_type(8)))  unsigned short  u16x8;

union FragU { v16bf v; u16x8 h[2]; };
union AccU  { v8f   v; float f[8]; };

#if __has_builtin(__builtin_amdgcn_tensor_load_to_lds) && __has_builtin(__builtin_amdgcn_s_wait_tensorcnt)
#define ATH_TDM 1
#else
#define ATH_TDM 0
#endif

__device__ __forceinline__ unsigned short f2b(float x) {
    unsigned u = __float_as_uint(x);
    u += 0x7fffu + ((u >> 16) & 1u);      // round-to-nearest-even
    return (unsigned short)(u >> 16);
}

__device__ __forceinline__ v8f wmma_bf16(v16bf a, v16bf b, v8f c) {
    // (neg_a, A, neg_b, B, c_mod, C, reuse_a, reuse_b)
    return __builtin_amdgcn_wmma_f32_16x16x32_bf16(false, a, false, b,
                                                   (short)0, c, false, false);
}

__device__ __forceinline__ v8f v8f_zero() {
    v8f z = {0.f, 0.f, 0.f, 0.f, 0.f, 0.f, 0.f, 0.f};
    return z;
}

#if ATH_TDM
typedef __attribute__((ext_vector_type(4))) unsigned tdm_uv4;
typedef __attribute__((ext_vector_type(8))) int      tdm_iv8;
typedef __attribute__((ext_vector_type(4))) int      tdm_iv4;

// Async DMA of a 2-D bf16 tile (tile0 elems wide x tile1 rows) from global
// [rows x stride0] into LDS (packed, row-major). Issued per-wave; completion
// tracked with TENSORcnt.  D# layout per CDNA5 ISA ch. 8 (group0 + group1).
__device__ __forceinline__ void tdm_load_tile_2d(const void* gsrc, void* lds_dst,
                                                 unsigned dim0, unsigned dim1,
                                                 unsigned tile0, unsigned tile1,
                                                 unsigned stride0) {
    unsigned long long ga = (unsigned long long)gsrc;
    // flat->LDS: low 32 bits of a flat shared pointer are the wave-relative
    // LDS byte offset (ISA 10.2 aperture mapping).
    unsigned lds_off = (unsigned)(unsigned long long)lds_dst;
    tdm_uv4 g0;
    g0[0] = 1u;                                               // count=1, user D#
    g0[1] = lds_off;                                          // lds_addr [63:32]
    g0[2] = (unsigned)ga;                                     // global_addr[31:0]
    g0[3] = (unsigned)((ga >> 32) & 0x01FFFFFFu) | (2u << 30);// addr[56:32] | type=2
    tdm_iv8 g1;
    g1[0] = (int)(1u << 16);                                  // data_size=1 (2B), wg_mask=0
    g1[1] = (int)((dim0 & 0xFFFFu) << 16);                    // tensor_dim0[15:0]
    g1[2] = (int)((dim0 >> 16) | ((dim1 & 0xFFFFu) << 16));   // dim0[31:16] | dim1[15:0]
    g1[3] = (int)((dim1 >> 16) | (tile0 << 16));              // dim1[31:16] | tile_dim0
    g1[4] = (int)(tile1 & 0xFFFFu);                           // tile_dim1 | tile_dim2=0
    g1[5] = (int)stride0;                                     // dim0_stride[31:0]
    g1[6] = 0;                                                // stride[47:32] | stride1 lo
    g1[7] = 0;
    tdm_iv4 z4 = {0, 0, 0, 0};                                // groups 2/3: tensor <= 2D
    tdm_iv8 z8 = {0, 0, 0, 0, 0, 0, 0, 0};
    __builtin_amdgcn_tensor_load_to_lds(g0, g1, z4, z4, z8, 0);
}
#endif

// ---------------------------------------------------------------------------
// fp32 -> bf16 conversion
// ---------------------------------------------------------------------------
__global__ void k_f32_to_bf16(const float* __restrict__ in,
                              unsigned short* __restrict__ out, int n) {
    int i = blockIdx.x * blockDim.x + threadIdx.x;
    if (i < n) out[i] = f2b(in[i]);
}

// conv weight [O=512, I=512, 3, 3] (OIHW)  ->  bf16 [O, t*512 + i], t = dy*3+dx
__global__ void k_reorder_convw(const float* __restrict__ w,
                                unsigned short* __restrict__ out) {
    int i = blockIdx.x * blockDim.x + threadIdx.x;
    if (i >= 512 * 4608) return;
    int o = i / 4608, k = i - o * 4608;
    int t = k >> 9, ic = k & 511;
    out[i] = f2b(w[(o * 512 + ic) * 9 + t]);
}

// ---------------------------------------------------------------------------
// Generic WMMA GEMM:  C[M,N] = A[M,K] * B[K,N] (A,B bf16 row-major, lda=K, ldb=N)
// block = 128 threads (4 waves), block tile 64x64, wave tile 32x32, K-step 32.
// A tile staged by the Tensor Data Mover (TENSORcnt); B staged transposed.
// mode 0: fp32 out [M,N], optional per-row divide (softmax normalization)
// mode 1: bf16 transposed out: out[n*ldt + m] = acc + bias[m]
// mode 2: like 1, plus squared copy at out[n*ldt + halfoff + m]   (V || V^2)
// ---------------------------------------------------------------------------
#define TP 40   // padded LDS row stride for Bs (elems)

__global__ void k_gemm_wmma(const unsigned short* __restrict__ A,
                            const unsigned short* __restrict__ B,
                            int M, int N, int K,
                            const float* __restrict__ bias,
                            const float* __restrict__ rowdiv,
                            void* __restrict__ outp,
                            int mode, int ldt, int halfoff) {
    __shared__ unsigned short As[64][32];   // packed: TDM writes tiles contiguously
    __shared__ unsigned short Bs[64][TP];   // stored transposed: Bs[n][k]

    const int tid  = threadIdx.x;
    const int lane = tid & 31, wave = tid >> 5;
    const int wr = wave >> 1, wc = wave & 1;
    const int l15 = lane & 15, hi = lane >> 4;
    const int mBase = blockIdx.y * 64;
    const int nBase = blockIdx.x * 64;

    v8f acc[2][2];
    acc[0][0] = v8f_zero(); acc[0][1] = v8f_zero();
    acc[1][0] = v8f_zero(); acc[1][1] = v8f_zero();

    for (int k0 = 0; k0 < K; k0 += 32) {
        // ---- stage A tile 64x32 via TDM (wave 0 issues the descriptor) ----
#if ATH_TDM
        if (wave == 0) {
            tdm_load_tile_2d(&A[(size_t)mBase * K + k0], &As[0][0],
                             (unsigned)K, (unsigned)(M - mBase),
                             32u, 64u, (unsigned)K);
        }
#else
#pragma unroll
        for (int j = 0; j < 2; ++j) {
            int v  = tid * 2 + j;
            int r  = v >> 2;
            int c8 = (v & 3) * 8;
            *(u16x8*)&As[r][c8] =
                *(const u16x8*)&A[(size_t)(mBase + r) * K + k0 + c8];
        }
#endif
        // ---- stage B tile 32x64, transposed into Bs[n][k] ----
#pragma unroll
        for (int j = 0; j < 2; ++j) {
            int v  = tid * 2 + j;
            int kk = v >> 3;
            int n8 = (v & 7) * 8;
            u16x8 g = *(const u16x8*)&B[(size_t)(k0 + kk) * N + nBase + n8];
#pragma unroll
            for (int e = 0; e < 8; ++e) Bs[n8 + e][kk] = g[e];
        }
#if ATH_TDM
        if (wave == 0) __builtin_amdgcn_s_wait_tensorcnt(0);
#endif
        __syncthreads();

        FragU a[2], b[2];
#pragma unroll
        for (int f = 0; f < 2; ++f) {
            const unsigned short* ap = &As[wr * 32 + f * 16 + l15][0];
            a[f].h[0] = *(const u16x8*)&ap[hi * 16];
            a[f].h[1] = *(const u16x8*)&ap[hi * 16 + 8];
            const unsigned short* bp = &Bs[wc * 32 + f * 16 + l15][0];
            b[f].h[0] = *(const u16x8*)&bp[hi * 16];
            b[f].h[1] = *(const u16x8*)&bp[hi * 16 + 8];
        }
#pragma unroll
        for (int f = 0; f < 2; ++f)
#pragma unroll
            for (int g = 0; g < 2; ++g)
                acc[f][g] = wmma_bf16(a[f].v, b[g].v, acc[f][g]);
        __syncthreads();
    }

    // ---- epilogue ----
#pragma unroll
    for (int f = 0; f < 2; ++f)
#pragma unroll
        for (int g = 0; g < 2; ++g) {
            AccU u; u.v = acc[f][g];
#pragma unroll
            for (int r = 0; r < 8; ++r) {
                int m = mBase + wr * 32 + f * 16 + r + 8 * hi;
                int n = nBase + wc * 32 + g * 16 + l15;
                float v = u.f[r];
                if (mode == 0) {
                    if (rowdiv) v /= rowdiv[m];
                    ((float*)outp)[(size_t)m * N + n] = v;
                } else {
                    if (bias) v += bias[m];
                    unsigned short* o = (unsigned short*)outp;
                    o[(size_t)n * ldt + m] = f2b(v);
                    if (mode == 2) o[(size_t)n * ldt + halfoff + m] = f2b(v * v);
                }
            }
        }
}

// ---------------------------------------------------------------------------
// Fused attention softmax: one wave handles 16 query rows.
//   Fq [Nc, Ck] bf16 (queries, row-major), Gt [Ns, Ck] bf16 (keys, transposed)
//   pass 1: row max of logits (WMMA, stats via C-tile <-> half-wave mapping)
//   pass 2: recompute, p = exp(l - max), write P [Nc, Ns] bf16, row sums
// ---------------------------------------------------------------------------
__global__ void k_attn_softmax(const unsigned short* __restrict__ Fq,
                               const unsigned short* __restrict__ Gt,
                               unsigned short* __restrict__ P,
                               float* __restrict__ rowsum,
                               int Ns, int Ck) {
    const int lane = threadIdx.x & 31;
    const int l15 = lane & 15, hi = lane >> 4;
    const int qb = blockIdx.x * 16;
    const unsigned short* qrow = Fq + (size_t)(qb + l15) * Ck;

    float mx[8];
#pragma unroll
    for (int r = 0; r < 8; ++r) mx[r] = -3.0e38f;

    for (int ns = 0; ns < Ns; ns += 16) {
        const unsigned short* krow = Gt + (size_t)(ns + l15) * Ck;
        if (ns + 16 < Ns) __builtin_prefetch(krow + 16 * Ck, 0, 0);
        v8f accv = v8f_zero();
        for (int k0 = 0; k0 < Ck; k0 += 32) {
            FragU a, b;
            a.h[0] = *(const u16x8*)&qrow[k0 + hi * 16];
            a.h[1] = *(const u16x8*)&qrow[k0 + hi * 16 + 8];
            b.h[0] = *(const u16x8*)&krow[k0 + hi * 16];
            b.h[1] = *(const u16x8*)&krow[k0 + hi * 16 + 8];
            accv = wmma_bf16(a.v, b.v, accv);
        }
        AccU u; u.v = accv;
#pragma unroll
        for (int r = 0; r < 8; ++r) mx[r] = fmaxf(mx[r], u.f[r]);
    }
#pragma unroll
    for (int r = 0; r < 8; ++r) {
#pragma unroll
        for (int off = 1; off < 16; off <<= 1)
            mx[r] = fmaxf(mx[r], __shfl_xor(mx[r], off, 32));
    }

    float sm[8];
#pragma unroll
    for (int r = 0; r < 8; ++r) sm[r] = 0.f;

    for (int ns = 0; ns < Ns; ns += 16) {
        const unsigned short* krow = Gt + (size_t)(ns + l15) * Ck;
        if (ns + 16 < Ns) __builtin_prefetch(krow + 16 * Ck, 0, 0);
        v8f accv = v8f_zero();
        for (int k0 = 0; k0 < Ck; k0 += 32) {
            FragU a, b;
            a.h[0] = *(const u16x8*)&qrow[k0 + hi * 16];
            a.h[1] = *(const u16x8*)&qrow[k0 + hi * 16 + 8];
            b.h[0] = *(const u16x8*)&krow[k0 + hi * 16];
            b.h[1] = *(const u16x8*)&krow[k0 + hi * 16 + 8];
            accv = wmma_bf16(a.v, b.v, accv);
        }
        AccU u; u.v = accv;
#pragma unroll
        for (int r = 0; r < 8; ++r) {
            float p = __expf(u.f[r] - mx[r]);
            sm[r] += p;
            P[(size_t)(qb + r + 8 * hi) * Ns + ns + l15] = f2b(p);
        }
    }
#pragma unroll
    for (int r = 0; r < 8; ++r) {
#pragma unroll
        for (int off = 1; off < 16; off <<= 1)
            sm[r] += __shfl_xor(sm[r], off, 32);
    }
    if (l15 == 0) {
#pragma unroll
        for (int r = 0; r < 8; ++r) rowsum[qb + r + 8 * hi] = sm[r];
    }
}

// ---------------------------------------------------------------------------
// Instance-norm per-channel stats: torch std(ddof=1) + 1e-6
// ---------------------------------------------------------------------------
__global__ void k_instnorm_stats(const float* __restrict__ x, int N,
                                 float* __restrict__ mu, float* __restrict__ isd) {
    const int c = blockIdx.x;
    const float* p = x + (size_t)c * N;
    float s = 0.f, s2 = 0.f;
    for (int i = threadIdx.x; i < N; i += blockDim.x) {
        float v = p[i]; s += v; s2 += v * v;
    }
    __shared__ float sh[256], sh2[256];
    sh[threadIdx.x] = s; sh2[threadIdx.x] = s2;
    __syncthreads();
    for (int o = 128; o > 0; o >>= 1) {
        if (threadIdx.x < o) {
            sh[threadIdx.x]  += sh[threadIdx.x + o];
            sh2[threadIdx.x] += sh2[threadIdx.x + o];
        }
        __syncthreads();
    }
    if (threadIdx.x == 0) {
        float m   = sh[0] / (float)N;
        float var = (sh2[0] - sh[0] * sh[0] / (float)N) / (float)(N - 1);
        var = fmaxf(var, 0.f);
        mu[c]  = m;
        isd[c] = 1.0f / (sqrtf(var) + 1e-6f);
    }
}

// out[c,n] = sqrt(relu(E[x^2]-E[x]^2)) * (x - mu_c) * isd_c + E[x]
// ME [N, 1024] fp32, cols 0..511 = mean, 512..1023 = E[x^2] (already /rowsum)
__global__ void k_adaattn_epilogue(const float* __restrict__ x,
                                   const float* __restrict__ ME,
                                   const float* __restrict__ mu,
                                   const float* __restrict__ isd,
                                   float* __restrict__ out, int N) {
    int i = blockIdx.x * blockDim.x + threadIdx.x;
    if (i >= 512 * N) return;
    int c = i / N, n = i - c * N;
    float me = ME[(size_t)n * 1024 + c];
    float e2 = ME[(size_t)n * 1024 + 512 + c];
    float sd = sqrtf(fmaxf(e2 - me * me, 0.f));
    out[i] = sd * (x[i] - mu[c]) * isd[c] + me;
}

// asum[512,128,128] += nearest-upsample2(a5[512,64,64])
__global__ void k_upsample_add(float* __restrict__ asum,
                               const float* __restrict__ a5) {
    int i = blockIdx.x * blockDim.x + threadIdx.x;
    if (i >= 512 * 16384) return;
    int c = i >> 14, n = i & 16383;
    int y = n >> 7, x = n & 127;
    asum[i] += a5[(size_t)c * 4096 + (y >> 1) * 64 + (x >> 1)];
}

// ---------------------------------------------------------------------------
// Implicit-GEMM 3x3 reflect-pad conv, 512->512, 128x128.
// A = reordered weights bf16 [512, 4608], B gathered on the fly from bf16 input.
// One wave per 16x16 output tile; K = 4608 (never straddles a (dy,dx) slot).
// ---------------------------------------------------------------------------
__global__ void k_conv3x3_wmma(const unsigned short* __restrict__ Wr,
                               const unsigned short* __restrict__ Xb,
                               const float* __restrict__ bias,
                               float* __restrict__ out) {
    const int lane = threadIdx.x & 31;
    const int l15 = lane & 15, hi = lane >> 4;
    const int NT = 16384 / 16;
    const int bid = blockIdx.x;
    const int m0 = (bid / NT) * 16;
    const int n0 = (bid % NT) * 16;
    const int n = n0 + l15;
    const int y = n >> 7, x = n & 127;
    const unsigned short* arow = Wr + (size_t)(m0 + l15) * 4608;

    v8f accv = v8f_zero();
    for (int k0 = 0; k0 < 4608; k0 += 32) {
        int t  = k0 >> 9;
        int dy = t / 3 - 1;
        int dx = t - (t / 3) * 3 - 1;
        int sy = y + dy; sy = sy < 0 ? -sy : (sy > 127 ? 254 - sy : sy);
        int sx = x + dx; sx = sx < 0 ? -sx : (sx > 127 ? 254 - sx : sx);
        int p  = sy * 128 + sx;
        int ib = (k0 & 511) + hi * 16;

        FragU a;
        a.h[0] = *(const u16x8*)&arow[k0 + hi * 16];
        a.h[1] = *(const u16x8*)&arow[k0 + hi * 16 + 8];
        union { v16bf v; unsigned short u[16]; } bb;
#pragma unroll
        for (int j = 0; j < 16; ++j)
            bb.u[j] = Xb[(size_t)(ib + j) * 16384 + p];

        accv = wmma_bf16(a.v, bb.v, accv);
    }
    AccU u; u.v = accv;
#pragma unroll
    for (int r = 0; r < 8; ++r) {
        int m = m0 + r + 8 * hi;
        out[(size_t)m * 16384 + n0 + l15] = u.f[r] + bias[m];
    }
}

// ---------------------------------------------------------------------------
// Host orchestration
// ---------------------------------------------------------------------------
static inline void cvt(hipStream_t s, const float* in, unsigned short* out, int n) {
    k_f32_to_bf16<<<(n + 255) / 256, 256, 0, s>>>(in, out, n);
}
static inline void gemm(hipStream_t s, const unsigned short* A, const unsigned short* B,
                        int M, int N, int K, const float* bias, const float* rowdiv,
                        void* out, int mode, int ldt, int halfoff) {
    dim3 g(N / 64, M / 64), b(128);
    k_gemm_wmma<<<g, b, 0, s>>>(A, B, M, N, K, bias, rowdiv, out, mode, ldt, halfoff);
}

extern "C" void kernel_launch(void* const* d_in, const int* in_sizes, int n_in,
                              void* d_out, int out_size, void* d_ws, size_t ws_size,
                              hipStream_t stream) {
    (void)in_sizes; (void)n_in; (void)out_size; (void)ws_size;
    const float* content4 = (const float*)d_in[0];
    const float* style4   = (const float*)d_in[1];
    const float* content5 = (const float*)d_in[2];
    const float* style5   = (const float*)d_in[3];
    const float* ckey4    = (const float*)d_in[4];
    const float* skey4    = (const float*)d_in[5];
    const float* ckey5    = (const float*)d_in[6];
    const float* skey5    = (const float*)d_in[7];
    const float* f4_w = (const float*)d_in[8];  const float* f4_b = (const float*)d_in[9];
    const float* g4_w = (const float*)d_in[10]; const float* g4_b = (const float*)d_in[11];
    const float* h4_w = (const float*)d_in[12]; const float* h4_b = (const float*)d_in[13];
    const float* f5_w = (const float*)d_in[14]; const float* f5_b = (const float*)d_in[15];
    const float* g5_w = (const float*)d_in[16]; const float* g5_b = (const float*)d_in[17];
    const float* h5_w = (const float*)d_in[18]; const float* h5_b = (const float*)d_in[19];
    const float* conv_w = (const float*)d_in[20];
    const float* conv_b = (const float*)d_in[21];

    const int N4 = 16384, Ns4 = 4096, Ck4 = 960;
    const int N5 = 4096,  Ns5 = 1024, Ck5 = 1472;

    // workspace bump allocator
    size_t off = 0;
    char* base = (char*)d_ws;
    auto alloc = [&](size_t bytes) -> char* {
        char* p = base + off;
        off += (bytes + 255) & ~(size_t)255;
        return p;
    };
    unsigned short* ck4b = (unsigned short*)alloc((size_t)Ck4 * N4 * 2);
    unsigned short* sk4b = (unsigned short*)alloc((size_t)Ck4 * Ns4 * 2);
    unsigned short* st4b = (unsigned short*)alloc((size_t)512 * Ns4 * 2);
    unsigned short* ck5b = (unsigned short*)alloc((size_t)Ck5 * N5 * 2);
    unsigned short* sk5b = (unsigned short*)alloc((size_t)Ck5 * Ns5 * 2);
    unsigned short* st5b = (unsigned short*)alloc((size_t)512 * Ns5 * 2);
    unsigned short* wf4  = (unsigned short*)alloc((size_t)Ck4 * Ck4 * 2);
    unsigned short* wg4  = (unsigned short*)alloc((size_t)Ck4 * Ck4 * 2);
    unsigned short* wh4  = (unsigned short*)alloc((size_t)512 * 512 * 2);
    unsigned short* wf5  = (unsigned short*)alloc((size_t)Ck5 * Ck5 * 2);
    unsigned short* wg5  = (unsigned short*)alloc((size_t)Ck5 * Ck5 * 2);
    unsigned short* wh5  = (unsigned short*)alloc((size_t)512 * 512 * 2);
    unsigned short* wcv  = (unsigned short*)alloc((size_t)512 * 4608 * 2);
    unsigned short* Fq4  = (unsigned short*)alloc((size_t)N4 * Ck4 * 2);
    unsigned short* Gt4  = (unsigned short*)alloc((size_t)Ns4 * Ck4 * 2);
    unsigned short* Vc4  = (unsigned short*)alloc((size_t)Ns4 * 1024 * 2);
    unsigned short* Fq5  = (unsigned short*)alloc((size_t)N5 * Ck5 * 2);
    unsigned short* Gt5  = (unsigned short*)alloc((size_t)Ns5 * Ck5 * 2);
    unsigned short* Vc5  = (unsigned short*)alloc((size_t)Ns5 * 1024 * 2);
    unsigned short* P4   = (unsigned short*)alloc((size_t)N4 * Ns4 * 2);
    unsigned short* P5   = (unsigned short*)alloc((size_t)N5 * Ns5 * 2);
    float* rs4  = (float*)alloc((size_t)N4 * 4);
    float* rs5  = (float*)alloc((size_t)N5 * 4);
    float* ME4  = (float*)alloc((size_t)N4 * 1024 * 4);
    float* ME5  = (float*)alloc((size_t)N5 * 1024 * 4);
    float* mu4  = (float*)alloc(512 * 4);
    float* isd4 = (float*)alloc(512 * 4);
    float* mu5  = (float*)alloc(512 * 4);
    float* isd5 = (float*)alloc(512 * 4);
    float* asum = (float*)alloc((size_t)512 * N4 * 4);
    float* a5b  = (float*)alloc((size_t)512 * N5 * 4);
    unsigned short* cb = (unsigned short*)alloc((size_t)512 * N4 * 2);

    // 1) bf16 conversions
    cvt(stream, ckey4, ck4b, Ck4 * N4);
    cvt(stream, skey4, sk4b, Ck4 * Ns4);
    cvt(stream, style4, st4b, 512 * Ns4);
    cvt(stream, ckey5, ck5b, Ck5 * N5);
    cvt(stream, skey5, sk5b, Ck5 * Ns5);
    cvt(stream, style5, st5b, 512 * Ns5);
    cvt(stream, f4_w, wf4, Ck4 * Ck4);
    cvt(stream, g4_w, wg4, Ck4 * Ck4);
    cvt(stream, h4_w, wh4, 512 * 512);
    cvt(stream, f5_w, wf5, Ck5 * Ck5);
    cvt(stream, g5_w, wg5, Ck5 * Ck5);
    cvt(stream, h5_w, wh5, 512 * 512);
    k_reorder_convw<<<(512 * 4608 + 255) / 256, 256, 0, stream>>>(conv_w, wcv);

    // 2) 1x1 conv GEMMs (transposed bf16 outputs; H -> V||V^2)
    gemm(stream, wf4, ck4b, Ck4, N4, Ck4, f4_b, nullptr, Fq4, 1, Ck4, 0);
    gemm(stream, wg4, sk4b, Ck4, Ns4, Ck4, g4_b, nullptr, Gt4, 1, Ck4, 0);
    gemm(stream, wh4, st4b, 512, Ns4, 512, h4_b, nullptr, Vc4, 2, 1024, 512);
    gemm(stream, wf5, ck5b, Ck5, N5, Ck5, f5_b, nullptr, Fq5, 1, Ck5, 0);
    gemm(stream, wg5, sk5b, Ck5, Ns5, Ck5, g5_b, nullptr, Gt5, 1, Ck5, 0);
    gemm(stream, wh5, st5b, 512, Ns5, 512, h5_b, nullptr, Vc5, 2, 1024, 512);

    // 3) attention softmax (P bf16 + row sums)
    k_attn_softmax<<<N4 / 16, 32, 0, stream>>>(Fq4, Gt4, P4, rs4, Ns4, Ck4);
    k_attn_softmax<<<N5 / 16, 32, 0, stream>>>(Fq5, Gt5, P5, rs5, Ns5, Ck5);

    // 4) P * [V||V^2], normalized by row sums -> [N, 1024] fp32 (mean | E[x^2])
    gemm(stream, P4, Vc4, N4, 1024, Ns4, nullptr, rs4, ME4, 0, 0, 0);
    gemm(stream, P5, Vc5, N5, 1024, Ns5, nullptr, rs5, ME5, 0, 0, 0);

    // 5) instance-norm stats + AdaAttN epilogues
    k_instnorm_stats<<<512, 256, 0, stream>>>(content4, N4, mu4, isd4);
    k_instnorm_stats<<<512, 256, 0, stream>>>(content5, N5, mu5, isd5);
    k_adaattn_epilogue<<<(512 * N4 + 255) / 256, 256, 0, stream>>>(content4, ME4, mu4, isd4, asum, N4);
    k_adaattn_epilogue<<<(512 * N5 + 255) / 256, 256, 0, stream>>>(content5, ME5, mu5, isd5, a5b, N5);

    // 6) asum += upsample2(a5); convert to bf16
    k_upsample_add<<<(512 * N4 + 255) / 256, 256, 0, stream>>>(asum, a5b);
    cvt(stream, asum, cb, 512 * N4);

    // 7) 3x3 reflect conv -> d_out
    k_conv3x3_wmma<<<(512 / 16) * (16384 / 16), 32, 0, stream>>>(wcv, cb, conv_b, (float*)d_out);
}
